// CrossModalAttention_43834436223633
// MI455X (gfx1250) — compile-verified, matrix operand order checked by reading, not run on hardware
//
#include <hip/hip_runtime.h>
#include <hip/hip_bf16.h>

#define DIM      768
#define HD       256
#define KT_STEPS 24              // 768 / 32
#define ST       784             // padded bf16 LDS row stride (elements): rows 32B-aligned
#define WELEM    (768 * 768)
#define LN_EPS   1e-5f
#define MT       4               // M-tiles per WG  -> M = 64 rows
#define NTW      3               // N-tiles per wave (16 waves x 3 = 48)
#define BLOCK    512

typedef __bf16 bf16_t;
typedef __attribute__((ext_vector_type(16))) __bf16 v16bf;
typedef __attribute__((ext_vector_type(8)))  float  v8f;

// ---- LDS: sQ/sCtx (aliased, 64 x ST bf16) + score table + LN stats ----------
#define SQ_BYTES     (64 * ST * 2)        // 100352
#define SCORE_FLOATS (64 * 3 * 3)         // 576
#define SMEM_BYTES   (SQ_BYTES + (SCORE_FLOATS + 64 + 64) * 4)

__device__ __forceinline__ v8f wmma_bf16(v16bf a, v16bf b, v8f c) {
  return __builtin_amdgcn_wmma_f32_16x16x32_bf16(false, a, false, b, (short)0, c,
                                                 false, false);
}

// Prepacked B fragment: [mat][nt][kt][lane][16 bf16], 32B per lane, contiguous.
__device__ __forceinline__ v16bf ld_b_frag(const bf16_t* W, int nt, int kt, int lane) {
  return *(const v16bf*)(W + ((((nt * KT_STEPS + kt) << 5) | lane) << 4));
}

// 16 consecutive fp32 at rowp+kbase -> bf16 A fragment.
__device__ __forceinline__ v16bf ld_a_global(const float* rowp, int kbase) {
  const float4* p = (const float4*)(rowp + kbase);
  float4 f0 = p[0], f1 = p[1], f2 = p[2], f3 = p[3];
  v16bf a;
  a[0]  = (bf16_t)f0.x; a[1]  = (bf16_t)f0.y; a[2]  = (bf16_t)f0.z; a[3]  = (bf16_t)f0.w;
  a[4]  = (bf16_t)f1.x; a[5]  = (bf16_t)f1.y; a[6]  = (bf16_t)f1.z; a[7]  = (bf16_t)f1.w;
  a[8]  = (bf16_t)f2.x; a[9]  = (bf16_t)f2.y; a[10] = (bf16_t)f2.z; a[11] = (bf16_t)f2.w;
  a[12] = (bf16_t)f3.x; a[13] = (bf16_t)f3.y; a[14] = (bf16_t)f3.z; a[15] = (bf16_t)f3.w;
  return a;
}

// Same, but rows pre-scaled by attention weights (per-lane scalars), up to 2 heads.
__device__ __forceinline__ void ld_a_scaled(const float* rowp, int kbase, float w0,
                                            float w1, bool two, v16bf& a0, v16bf& a1) {
  const float4* p = (const float4*)(rowp + kbase);
  float4 f0 = p[0], f1 = p[1], f2 = p[2], f3 = p[3];
  float v[16] = {f0.x, f0.y, f0.z, f0.w, f1.x, f1.y, f1.z, f1.w,
                 f2.x, f2.y, f2.z, f2.w, f3.x, f3.y, f3.z, f3.w};
#pragma unroll
  for (int i = 0; i < 16; ++i) a0[i] = (bf16_t)(v[i] * w0);
  if (two) {
#pragma unroll
    for (int i = 0; i < 16; ++i) a1[i] = (bf16_t)(v[i] * w1);
  } else {
    a1 = a0;
  }
}

// ---- Pre-pass: fp32 weights -> bf16 fragment-major layout in workspace ------
__global__ __launch_bounds__(256) void prepack_weights_kernel(
    const float* __restrict__ Wq, const float* __restrict__ Wk,
    const float* __restrict__ Wv, const float* __restrict__ Wo,
    bf16_t* __restrict__ wpk) {
  int i = blockIdx.x * 256 + threadIdx.x;
  const int TOT = 4 * 48 * KT_STEPS * 32 * 16;   // 2359296
  if (i >= TOT) return;
  int slot = i & 15;
  int r = i >> 4;
  int lane = r & 31; r >>= 5;
  int kt = r % KT_STEPS; r /= KT_STEPS;
  int nt = r % 48;
  int mat = r / 48;
  const float* W = (mat == 0) ? Wq : (mat == 1) ? Wk : (mat == 2) ? Wv : Wo;
  int n = (nt << 4) + (lane & 15);
  int k = (kt << 5) + ((lane >> 4) << 4) + slot;
  wpk[i] = (bf16_t)W[n * DIM + k];
}

// ---- Fused attention block: 1 WG (512 thr, 16 waves) = 64 batch rows --------
__global__ __launch_bounds__(BLOCK, 1) void fused_xattn_ln_kernel(
    const float* __restrict__ query, const float* __restrict__ key0,
    const float* __restrict__ key1, const float* __restrict__ key2,
    const bf16_t* __restrict__ wpk,
    const float* __restrict__ bq, const float* __restrict__ bk,
    const float* __restrict__ bv, const float* __restrict__ bo,
    const float* __restrict__ gamma, const float* __restrict__ beta,
    float* __restrict__ out) {
  __shared__ __align__(32) unsigned char smem[SMEM_BYTES];
  bf16_t* sQ    = (bf16_t*)smem;                    // 64 x ST bf16
  bf16_t* sCtx  = sQ;                               // alias: sQ dead after scores
  float*  sScore = (float*)(smem + SQ_BYTES);       // [64][3][3] scores -> attn
  float*  sSum   = sScore + SCORE_FLOATS;           // [64]
  float*  sSsq   = sSum + 64;                       // [64]

  const int tid  = threadIdx.x;
  const int wave = tid >> 5;
  const int lane = tid & 31;
  const int mrow  = lane & 15;          // A-fragment row within M-tile
  const int khalf = (lane >> 4) << 4;   // A-fragment K offset (0/16)
  const int nc    = lane & 15;          // C/D fragment column-in-tile
  const int mo    = (lane >> 4) << 3;   // C/D fragment row offset (0/8)
  const int row0  = blockIdx.x << 6;    // 64 rows per WG
  const int nt0   = wave * NTW;

  int ncol[NTW], hh[NTW];
#pragma unroll
  for (int j = 0; j < NTW; ++j) {
    ncol[j] = ((nt0 + j) << 4) + nc;
    hh[j]   = (nt0 + j) >> 4;           // head of this N-tile (256 cols/head)
  }
  const bool twoHeads = (hh[NTW - 1] != hh[0]);

  const bf16_t* WqP = wpk;
  const bf16_t* WkP = wpk + WELEM;
  const bf16_t* WvP = wpk + 2 * WELEM;
  const bf16_t* WoP = wpk + 3 * WELEM;
  const float* kvp[3] = {key0, key1, key2};

  // zero score + stats accumulators
  for (int i = tid; i < SCORE_FLOATS + 128; i += BLOCK) sScore[i] = 0.0f;

  v8f acc[MT][NTW];

  // ---------------- GEMM1: Q = query @ Wq^T + bq -> sQ (bf16) ----------------
  {
#pragma unroll
    for (int mt = 0; mt < MT; ++mt)
#pragma unroll
      for (int j = 0; j < NTW; ++j)
#pragma unroll
        for (int r = 0; r < 8; ++r) acc[mt][j][r] = 0.0f;
    const float* rp[MT];
#pragma unroll
    for (int mt = 0; mt < MT; ++mt)
      rp[mt] = query + (size_t)(row0 + mt * 16 + mrow) * DIM;
#pragma unroll 2
    for (int kt = 0; kt < KT_STEPS; ++kt) {
      v16bf b[NTW];
#pragma unroll
      for (int j = 0; j < NTW; ++j) b[j] = ld_b_frag(WqP, nt0 + j, kt, lane);
      __builtin_prefetch(WqP + ((((nt0 * KT_STEPS + kt + 2) << 5) | lane) << 4), 0, 0);
#pragma unroll
      for (int mt = 0; mt < MT; ++mt) {
        v16bf a = ld_a_global(rp[mt], (kt << 5) + khalf);
#pragma unroll
        for (int j = 0; j < NTW; ++j) acc[mt][j] = wmma_bf16(a, b[j], acc[mt][j]);
      }
    }
#pragma unroll
    for (int j = 0; j < NTW; ++j) {
      float bias = bq[ncol[j]];
#pragma unroll
      for (int mt = 0; mt < MT; ++mt)
#pragma unroll
        for (int r = 0; r < 8; ++r)
          sQ[(mt * 16 + r + mo) * ST + ncol[j]] = (bf16_t)(acc[mt][j][r] + bias);
    }
  }
  __syncthreads();

  // ------ GEMM2: K_s = key_s @ Wk^T + bk, fused scores += Q .* K_s -----------
  for (int s = 0; s < 3; ++s) {
#pragma unroll
    for (int mt = 0; mt < MT; ++mt)
#pragma unroll
      for (int j = 0; j < NTW; ++j)
#pragma unroll
        for (int r = 0; r < 8; ++r) acc[mt][j][r] = 0.0f;
    const float* rp[MT];
#pragma unroll
    for (int mt = 0; mt < MT; ++mt)
      rp[mt] = kvp[s] + (size_t)(row0 + mt * 16 + mrow) * DIM;
#pragma unroll 2
    for (int kt = 0; kt < KT_STEPS; ++kt) {
      v16bf b[NTW];
#pragma unroll
      for (int j = 0; j < NTW; ++j) b[j] = ld_b_frag(WkP, nt0 + j, kt, lane);
#pragma unroll
      for (int mt = 0; mt < MT; ++mt) {
        v16bf a = ld_a_global(rp[mt], (kt << 5) + khalf);
#pragma unroll
        for (int j = 0; j < NTW; ++j) acc[mt][j] = wmma_bf16(a, b[j], acc[mt][j]);
      }
    }
    // epilogue: per-row, per-head partial dot(Q, K_s) -> LDS atomic accumulate
#pragma unroll
    for (int mt = 0; mt < MT; ++mt) {
#pragma unroll
      for (int r = 0; r < 8; ++r) {
        int gm = mt * 16 + r + mo;
        float pl = 0.0f, ph = 0.0f;
#pragma unroll
        for (int j = 0; j < NTW; ++j) {
          float kvv = acc[mt][j][r] + bk[ncol[j]];
          float qv  = (float)sQ[gm * ST + ncol[j]];
          float p   = qv * kvv;
          if (hh[j] == hh[0]) pl += p; else ph += p;
        }
        atomicAdd(&sScore[(gm * 3 + hh[0]) * 3 + s], pl);
        if (twoHeads) atomicAdd(&sScore[(gm * 3 + hh[NTW - 1]) * 3 + s], ph);
      }
    }
  }
  __syncthreads();

  // ---------------- softmax over S=3 (scale 1/sqrt(256)) --------------------
  if (tid < 192) {
    float* p = &sScore[tid * 3];
    float s0 = p[0] * (1.0f / 16.0f), s1 = p[1] * (1.0f / 16.0f), s2 = p[2] * (1.0f / 16.0f);
    float mx = fmaxf(s0, fmaxf(s1, s2));
    float e0 = __expf(s0 - mx), e1 = __expf(s1 - mx), e2 = __expf(s2 - mx);
    float inv = 1.0f / (e0 + e1 + e2);
    p[0] = e0 * inv; p[1] = e1 * inv; p[2] = e2 * inv;
  }
  __syncthreads();

  // ---- GEMM3: ctx = sum_s (attn_s .* key_s) @ Wv^T  (A rows pre-scaled) -----
  {
    v8f ctx[MT][NTW];
#pragma unroll
    for (int mt = 0; mt < MT; ++mt)
#pragma unroll
      for (int j = 0; j < NTW; ++j)
#pragma unroll
        for (int r = 0; r < 8; ++r) ctx[mt][j][r] = 0.0f;
    for (int s = 0; s < 3; ++s) {
      const float* rp[MT];
      float wLo[MT], wHi[MT];
#pragma unroll
      for (int mt = 0; mt < MT; ++mt) {
        rp[mt] = kvp[s] + (size_t)(row0 + mt * 16 + mrow) * DIM;
        int gm = mt * 16 + mrow;                 // A-fragment row for this lane
        wLo[mt] = sScore[(gm * 3 + hh[0]) * 3 + s];
        wHi[mt] = sScore[(gm * 3 + hh[NTW - 1]) * 3 + s];
      }
#pragma unroll 2
      for (int kt = 0; kt < KT_STEPS; ++kt) {
        v16bf b[NTW];
#pragma unroll
        for (int j = 0; j < NTW; ++j) b[j] = ld_b_frag(WvP, nt0 + j, kt, lane);
#pragma unroll
        for (int mt = 0; mt < MT; ++mt) {
          v16bf aLo, aHi;
          ld_a_scaled(rp[mt], (kt << 5) + khalf, wLo[mt], wHi[mt], twoHeads, aLo, aHi);
#pragma unroll
          for (int j = 0; j < NTW; ++j)
            ctx[mt][j] = wmma_bf16((hh[j] == hh[0]) ? aLo : aHi, b[j], ctx[mt][j]);
        }
      }
    }
    // + bv (softmax weights sum to 1) -> sCtx (bf16)
#pragma unroll
    for (int j = 0; j < NTW; ++j) {
      float bias = bv[ncol[j]];
#pragma unroll
      for (int mt = 0; mt < MT; ++mt)
#pragma unroll
        for (int r = 0; r < 8; ++r)
          sCtx[(mt * 16 + r + mo) * ST + ncol[j]] = (bf16_t)(ctx[mt][j][r] + bias);
    }
  }
  __syncthreads();

  // ------ GEMM4: x = ctx @ Wo^T + bo + query; LN stats via LDS partials ------
  {
#pragma unroll
    for (int mt = 0; mt < MT; ++mt)
#pragma unroll
      for (int j = 0; j < NTW; ++j)
#pragma unroll
        for (int r = 0; r < 8; ++r) acc[mt][j][r] = 0.0f;
#pragma unroll 2
    for (int kt = 0; kt < KT_STEPS; ++kt) {
      v16bf b[NTW];
#pragma unroll
      for (int j = 0; j < NTW; ++j) b[j] = ld_b_frag(WoP, nt0 + j, kt, lane);
#pragma unroll
      for (int mt = 0; mt < MT; ++mt) {
        v16bf a = *(const v16bf*)(sCtx + (mt * 16 + mrow) * ST + (kt << 5) + khalf);
#pragma unroll
        for (int j = 0; j < NTW; ++j) acc[mt][j] = wmma_bf16(a, b[j], acc[mt][j]);
      }
    }
    // x = acc + bo + residual (kept in registers through the stats barrier)
#pragma unroll
    for (int j = 0; j < NTW; ++j) {
      float bias = bo[ncol[j]];
#pragma unroll
      for (int mt = 0; mt < MT; ++mt)
#pragma unroll
        for (int r = 0; r < 8; ++r) {
          int gm = mt * 16 + r + mo;
          acc[mt][j][r] += bias + query[(size_t)(row0 + gm) * DIM + ncol[j]];
        }
    }
#pragma unroll
    for (int mt = 0; mt < MT; ++mt)
#pragma unroll
      for (int r = 0; r < 8; ++r) {
        int gm = mt * 16 + r + mo;
        float s1 = 0.0f, s2 = 0.0f;
#pragma unroll
        for (int j = 0; j < NTW; ++j) {
          float x = acc[mt][j][r];
          s1 += x; s2 += x * x;
        }
        atomicAdd(&sSum[gm], s1);
        atomicAdd(&sSsq[gm], s2);
      }
  }
  __syncthreads();
  if (tid < 64) {
    float mean = sSum[tid] * (1.0f / 768.0f);
    float var  = sSsq[tid] * (1.0f / 768.0f) - mean * mean;
    sSum[tid] = mean;
    sSsq[tid] = rsqrtf(var + LN_EPS);
  }
  __syncthreads();

  // ---------------- normalize + store ---------------------------------------
  {
    float g[NTW], be[NTW];
#pragma unroll
    for (int j = 0; j < NTW; ++j) { g[j] = gamma[ncol[j]]; be[j] = beta[ncol[j]]; }
#pragma unroll
    for (int mt = 0; mt < MT; ++mt)
#pragma unroll
      for (int r = 0; r < 8; ++r) {
        int gm = mt * 16 + r + mo;
        float mean = sSum[gm], rstd = sSsq[gm];
#pragma unroll
        for (int j = 0; j < NTW; ++j)
          out[(size_t)(row0 + gm) * DIM + ncol[j]] =
              (acc[mt][j][r] - mean) * rstd * g[j] + be[j];
      }
  }
}

extern "C" void kernel_launch(void* const* d_in, const int* in_sizes, int n_in,
                              void* d_out, int out_size, void* d_ws, size_t ws_size,
                              hipStream_t stream) {
  const float* query = (const float*)d_in[0];
  const float* key0  = (const float*)d_in[1];
  const float* key1  = (const float*)d_in[2];
  const float* key2  = (const float*)d_in[3];
  const float* Wq    = (const float*)d_in[4];
  const float* bq    = (const float*)d_in[5];
  const float* Wk    = (const float*)d_in[6];
  const float* bk    = (const float*)d_in[7];
  const float* Wv    = (const float*)d_in[8];
  const float* bv    = (const float*)d_in[9];
  const float* Wo    = (const float*)d_in[10];
  const float* bo    = (const float*)d_in[11];
  const float* gamma = (const float*)d_in[12];
  const float* beta  = (const float*)d_in[13];

  bf16_t* wpk = (bf16_t*)d_ws;                 // 4 * 589824 bf16 = 4.5 MB
  const int B = in_sizes[0] / DIM;             // 65536

  prepack_weights_kernel<<<(4 * WELEM + 255) / 256, 256, 0, stream>>>(
      Wq, Wk, Wv, Wo, wpk);
  fused_xattn_ln_kernel<<<B / 64, BLOCK, 0, stream>>>(
      query, key0, key1, key2, wpk, bq, bk, bv, bo, gamma, beta, (float*)d_out);
}
